// MAE_Encoder_67465346286039
// MI455X (gfx1250) — matrix operand order repaired and checked by reading, not compiled, not run on hardware
//
#include <hip/hip_runtime.h>
#include <hip/hip_bf16.h>

// MAE-ViT encoder (B=128, T'=50 tokens, C=384, NH=6, L=12, MoE E=8 top-2 dense combine)
// Large GEMMs: v_wmma_f32_16x16x32_bf16 (bf16 in, f32 accum).
// A-tile staged with global_load_async_to_lds_b128 (ASYNCcnt path);
// W-tile f32->bf16 converted in-register, packed ds_store_b64;
// epilogue staged through LDS for coalesced b128 global traffic.

typedef __attribute__((ext_vector_type(16))) __bf16 v16bf;
typedef __attribute__((ext_vector_type(8)))  float  v8f;

#define Bc   128
#define Tt   50          // RT+1
#define NTOK (Bc*Tt)     // 6400
#define Cc   384
#define NHh  6
#define HDd  64
#define HIDc 1536
#define Ee   8
#define Tfull 196

__device__ __forceinline__ unsigned short f2bf(float f) {
    unsigned int u = __builtin_bit_cast(unsigned int, f);
    unsigned int r = u + 0x7FFFu + ((u >> 16) & 1u);   // round-to-nearest-even
    return (unsigned short)(r >> 16);
}
__device__ __forceinline__ float bf2f(unsigned short b) {
    unsigned int u = ((unsigned int)b) << 16;
    return __builtin_bit_cast(float, u);
}
__device__ __forceinline__ float gelu_exact(float z) {
    return 0.5f * z * (1.0f + erff(z * 0.70710678118654752f));
}

// ---------------------------------------------------------------- GEMM (WMMA)
// out[M,N] = f(A_bf16[M,K] @ W_f32[K,N] + bias), M = NTOK (multiple of 128),
// N mult of 64, K mult of 32.  Block tile 128x64, 8 waves -> each wave 32x32.
enum { M_STORE = 0, M_ACC_BIAS = 1, M_GELU_BF16 = 2, M_ACC_BIAS_SCALE = 3 };

template <int MODE>
__global__ __launch_bounds__(256)
void gemm_wmma(const unsigned short* __restrict__ A, const float* __restrict__ W,
               const float* __restrict__ bias, float* __restrict__ outF,
               unsigned short* __restrict__ outB, const float* __restrict__ rowscale,
               int K, int NC)
{
    constexpr int BM = 128, BN = 64, BK = 32, SK = 40; // SK: padded row (halves), 80B
    union SMem {
        struct { unsigned short a[BM * SK]; unsigned short b[BN * SK]; } ab;
        float c[BM * BN];                               // epilogue staging (32 KB)
    };
    __shared__ __align__(16) SMem sm;

    const int tid  = threadIdx.x;
    const int lane = tid & 31;
    const int wave = tid >> 5;
    const int wm = wave >> 1, wn = wave & 1;     // 4 x 2 wave grid
    const int m0 = blockIdx.y * BM;
    const int n0 = blockIdx.x * BN;
    const int hs = lane >> 4;                    // half-wave select
    const int lr = lane & 15;

    v8f acc[2][2];
    for (int s = 0; s < 2; ++s)
        for (int t = 0; t < 2; ++t) acc[s][t] = (v8f)0.0f;

    // A staging: each thread owns 32 contiguous bytes of one row
    const int arow = tid >> 1;                   // A tile: 128 rows x 32 cols
    const int acol = (tid & 1) * 16;
    const unsigned short* gA = A + (size_t)(m0 + arow) * K + acol;
    const unsigned ldsA = (unsigned)(uintptr_t)(&sm.ab.a[arow * SK + acol]);

    // W staging: thread owns (n, k-group of 4) -> one packed b64 LDS store
    const int wnn = tid & 63;
    const int wk0 = (tid >> 6) * 4;              // 0,4,8,12
    const float* wp0 = W + (size_t)wk0 * NC + n0 + wnn;
    const float* wp1 = wp0 + (size_t)16 * NC;
    unsigned short* const sbp0 = &sm.ab.b[wnn * SK + wk0];
    unsigned short* const sbp1 = sbp0 + 16;

    for (int k0 = 0; k0 < K; k0 += BK) {
        // async copy A tile straight to LDS (no VGPR round-trip; ASYNCcnt)
        asm volatile("global_load_async_to_lds_b128 %0, %1, off"
                     :: "v"(ldsA), "v"((unsigned long long)gA) : "memory");
        asm volatile("global_load_async_to_lds_b128 %0, %1, off offset:16"
                     :: "v"(ldsA), "v"((unsigned long long)gA) : "memory");
        // stage W (f32 -> bf16) transposed; 8 coalesced b32 loads, 2 b64 LDS stores
        ushort4 p0, p1;
        p0.x = f2bf(wp0[0]);
        p0.y = f2bf(wp0[(size_t)NC]);
        p0.z = f2bf(wp0[(size_t)2 * NC]);
        p0.w = f2bf(wp0[(size_t)3 * NC]);
        p1.x = f2bf(wp1[0]);
        p1.y = f2bf(wp1[(size_t)NC]);
        p1.z = f2bf(wp1[(size_t)2 * NC]);
        p1.w = f2bf(wp1[(size_t)3 * NC]);
        *reinterpret_cast<ushort4*>(sbp0) = p0;
        *reinterpret_cast<ushort4*>(sbp1) = p1;
        if (k0 + BK < K) {                       // global_prefetch_b8 next K-tile
            __builtin_prefetch(gA + BK, 0, 1);
            __builtin_prefetch(wp0 + (size_t)BK * NC, 0, 1);
            __builtin_prefetch(wp1 + (size_t)BK * NC, 0, 1);
        }
        gA  += BK;
        wp0 += (size_t)BK * NC;
        wp1 += (size_t)BK * NC;
        asm volatile("s_wait_asynccnt 0x0" ::: "memory");
        __syncthreads();

        v16bf afr[2], bfr[2];
        // A 16x32 bf16 fragment: lanes<16 hold K0..7,K16..23; lanes>=16 hold K8..15,K24..31
        for (int s = 0; s < 2; ++s) {
            int row = wm * 32 + s * 16 + lr;
            union { uint4 q[2]; v16bf v; } u;
            u.q[0] = *reinterpret_cast<const uint4*>(&sm.ab.a[row * SK + 8 * hs]);
            u.q[1] = *reinterpret_cast<const uint4*>(&sm.ab.a[row * SK + 16 + 8 * hs]);
            afr[s] = u.v;
        }
        // B 32x16 bf16 fragment: lane n<16 -> col n, K0..7,K16..23 ; lane n+16 -> col n, K8..15,K24..31
        for (int t = 0; t < 2; ++t) {
            int col = wn * 32 + t * 16 + lr;
            union { uint4 q[2]; v16bf v; } u;
            u.q[0] = *reinterpret_cast<const uint4*>(&sm.ab.b[col * SK + 8 * hs]);
            u.q[1] = *reinterpret_cast<const uint4*>(&sm.ab.b[col * SK + 16 + 8 * hs]);
            bfr[t] = u.v;
        }
        for (int s = 0; s < 2; ++s)
            for (int t = 0; t < 2; ++t)
                acc[s][t] = __builtin_amdgcn_wmma_f32_16x16x32_bf16(
                    false, afr[s], false, bfr[t], (short)0, acc[s][t], false, false);
        __syncthreads();
    }

    // ---- epilogue: accumulators -> LDS tile -> coalesced b128 global traffic
    // D layout => lane<16: N=lane, M=r ; lane>=16: N=lane-16, M=r+8
    for (int s = 0; s < 2; ++s)
        for (int t = 0; t < 2; ++t) {
            int lcol = wn * 32 + t * 16 + lr;
            int lrow0 = wm * 32 + s * 16 + hs * 8;
            for (int r = 0; r < 8; ++r)
                sm.c[(lrow0 + r) * BN + lcol] = acc[s][t][r];
        }
    __syncthreads();

    const int cg   = tid & 15;                   // column group: 4 floats
    const int rsub = tid >> 4;                   // 16 rows per pass
    const int colb = n0 + cg * 4;
    float b0 = 0.f, b1v = 0.f, b2v = 0.f, b3 = 0.f;
    if (MODE != M_STORE) {
        b0 = bias[colb]; b1v = bias[colb + 1]; b2v = bias[colb + 2]; b3 = bias[colb + 3];
    }
    for (int p = 0; p < 8; ++p) {
        int row  = p * 16 + rsub;
        int grow = m0 + row;
        size_t idx = (size_t)grow * NC + colb;
        const float* cp = &sm.c[row * BN + cg * 4];
        float v0 = cp[0], v1 = cp[1], v2 = cp[2], v3 = cp[3];
        if (MODE == M_STORE) {
            float4 o = {v0, v1, v2, v3};
            *reinterpret_cast<float4*>(outF + idx) = o;
        } else if (MODE == M_ACC_BIAS) {
            float4 g = *reinterpret_cast<const float4*>(outF + idx);
            float4 o = {g.x + v0 + b0, g.y + v1 + b1v, g.z + v2 + b2v, g.w + v3 + b3};
            *reinterpret_cast<float4*>(outF + idx) = o;
        } else if (MODE == M_GELU_BF16) {
            ushort4 o;
            o.x = f2bf(gelu_exact(v0 + b0));
            o.y = f2bf(gelu_exact(v1 + b1v));
            o.z = f2bf(gelu_exact(v2 + b2v));
            o.w = f2bf(gelu_exact(v3 + b3));
            *reinterpret_cast<ushort4*>(outB + idx) = o;
        } else {                                   // MoE dense combine
            float sc = rowscale[(size_t)grow * Ee];
            float4 g = *reinterpret_cast<const float4*>(outF + idx);
            float4 o = {g.x + sc * (v0 + b0), g.y + sc * (v1 + b1v),
                        g.z + sc * (v2 + b2v), g.w + sc * (v3 + b3)};
            *reinterpret_cast<float4*>(outF + idx) = o;
        }
    }
}

// ---------------------------------------------------------------- LayerNorm
__global__ __launch_bounds__(256)
void ln_kernel(const float* __restrict__ x, const float* __restrict__ g,
               const float* __restrict__ b, unsigned short* __restrict__ out)
{
    int tok  = blockIdx.x * 8 + (threadIdx.x >> 5);  // one wave32 per token
    int lane = threadIdx.x & 31;
    const float* xr = x + (size_t)tok * Cc;
    float s = 0.f, sq = 0.f, v[12];
    for (int i = 0; i < 12; ++i) { float t = xr[lane + 32 * i]; v[i] = t; s += t; sq += t * t; }
    for (int off = 16; off; off >>= 1) { s += __shfl_xor(s, off, 32); sq += __shfl_xor(sq, off, 32); }
    float mean = s * (1.f / Cc), var = sq * (1.f / Cc) - mean * mean;
    float rs = rsqrtf(var + 1e-5f);
    for (int i = 0; i < 12; ++i) {
        int c = lane + 32 * i;
        out[(size_t)tok * Cc + c] = f2bf((v[i] - mean) * rs * g[c] + b[c]);
    }
}

// final LN: f32 out, transposed to [T', B, C]
__global__ __launch_bounds__(256)
void lnf_kernel(const float* __restrict__ x, const float* __restrict__ g,
                const float* __restrict__ b, float* __restrict__ out)
{
    int tok  = blockIdx.x * 8 + (threadIdx.x >> 5);
    int lane = threadIdx.x & 31;
    const float* xr = x + (size_t)tok * Cc;
    float s = 0.f, sq = 0.f, v[12];
    for (int i = 0; i < 12; ++i) { float t = xr[lane + 32 * i]; v[i] = t; s += t; sq += t * t; }
    for (int off = 16; off; off >>= 1) { s += __shfl_xor(s, off, 32); sq += __shfl_xor(sq, off, 32); }
    float mean = s * (1.f / Cc), var = sq * (1.f / Cc) - mean * mean;
    float rs = rsqrtf(var + 1e-5f);
    int bb = tok / Tt, tt = tok % Tt;
    float* orow = out + ((size_t)tt * Bc + bb) * Cc;
    for (int i = 0; i < 12; ++i) {
        int c = lane + 32 * i;
        orow[c] = (v[i] - mean) * rs * g[c] + b[c];
    }
}

// ---------------------------------------------------------------- Attention
// one workgroup per (batch, head); seq=50, hd=64; all in LDS
__global__ __launch_bounds__(256)
void attn_kernel(const float* __restrict__ qkv, unsigned short* __restrict__ o)
{
    __shared__ float sq[Tt * HDd], sk[Tt * HDd], sv[Tt * HDd], sa[Tt * Tt];
    int b = blockIdx.x / NHh, h = blockIdx.x % NHh;
    const float* base = qkv + (size_t)b * Tt * (3 * Cc) + h * HDd;
    for (int idx = threadIdx.x; idx < Tt * HDd; idx += 256) {
        int t = idx >> 6, d = idx & 63;
        const float* row = base + (size_t)t * (3 * Cc);
        sq[idx] = row[d]; sk[idx] = row[Cc + d]; sv[idx] = row[2 * Cc + d];
    }
    __syncthreads();
    for (int e = threadIdx.x; e < Tt * Tt; e += 256) {
        int i = e / Tt, j = e % Tt;
        const float* qi = sq + i * HDd; const float* kj = sk + j * HDd;
        float acc = 0.f;
        for (int d = 0; d < HDd; ++d) acc += qi[d] * kj[d];
        sa[e] = acc * 0.125f;                         // HD^-0.5
    }
    __syncthreads();
    if (threadIdx.x < Tt) {                           // row softmax
        float* r = sa + threadIdx.x * Tt;
        float mx = r[0];
        for (int j = 1; j < Tt; ++j) mx = fmaxf(mx, r[j]);
        float s = 0.f;
        for (int j = 0; j < Tt; ++j) { float ev = __expf(r[j] - mx); r[j] = ev; s += ev; }
        float inv = 1.f / s;
        for (int j = 0; j < Tt; ++j) r[j] *= inv;
    }
    __syncthreads();
    for (int e = threadIdx.x; e < Tt * HDd; e += 256) {
        int i = e >> 6, d = e & 63;
        const float* ar = sa + i * Tt;
        float acc = 0.f;
        for (int j = 0; j < Tt; ++j) acc += ar[j] * sv[j * HDd + d];
        o[((size_t)b * Tt + i) * Cc + h * HDd + d] = f2bf(acc);
    }
}

// ---------------------------------------------------------------- MoE gate
__global__ __launch_bounds__(256)
void gate_kernel(const unsigned short* __restrict__ f, const float* __restrict__ gw,
                 const float* __restrict__ gb, float* __restrict__ we)
{
    int tok = blockIdx.x * 256 + threadIdx.x;        // 6400 threads
    float lg[Ee];
    for (int e = 0; e < Ee; ++e) lg[e] = gb[e];
    const unsigned short* fr = f + (size_t)tok * Cc;
    for (int c = 0; c < Cc; ++c) {
        float fv = bf2f(fr[c]);
        for (int e = 0; e < Ee; ++e) lg[e] += fv * gw[c * Ee + e];
    }
    int i1 = 0;
    for (int e = 1; e < Ee; ++e) if (lg[e] > lg[i1]) i1 = e;   // stable: first wins
    int i2 = -1;
    for (int e = 0; e < Ee; ++e) {
        if (e == i1) continue;
        if (i2 < 0 || lg[e] > lg[i2]) i2 = e;
    }
    float e2  = __expf(lg[i2] - lg[i1]);             // softmax over top-2
    float inv = 1.f / (1.f + e2);
    float* wr = we + (size_t)tok * Ee;
    for (int e = 0; e < Ee; ++e) wr[e] = 0.f;
    wr[i1] = inv;
    wr[i2] = e2 * inv;
}

// ---------------------------------------------------------------- Patchify
// workgroup per (b, out-token); token0 = cls; token i>0 <- shuffled patch
__global__ __launch_bounds__(256)
void patchify_kernel(const float* __restrict__ img, const int* __restrict__ fwd,
                     const float* __restrict__ cw, const float* __restrict__ cb,
                     const float* __restrict__ pos, const float* __restrict__ cls,
                     float* __restrict__ x)
{
    __shared__ float sp[768];
    int b = blockIdx.x / Tt, i = blockIdx.x % Tt;
    float* xr = x + ((size_t)b * Tt + i) * Cc;
    if (i == 0) {
        for (int c = threadIdx.x; c < Cc; c += 256) xr[c] = cls[c];
        return;
    }
    int t = fwd[(size_t)(i - 1) * Bc + b];
    int ph = t / 14, pw = t % 14;
    for (int idx = threadIdx.x; idx < 768; idx += 256) {
        int ci = idx >> 8, rem = idx & 255, py = rem >> 4, px = rem & 15;
        sp[idx] = img[(((size_t)b * 3 + ci) * 224 + ph * 16 + py) * 224 + pw * 16 + px];
    }
    __syncthreads();
    for (int c = threadIdx.x; c < Cc; c += 256) {
        float acc = cb[c] + pos[(size_t)t * Cc + c];
        const float* wr = cw + (size_t)c * 768;      // conv_w [C,3,16,16] matches sp order
        for (int k = 0; k < 768; ++k) acc += sp[k] * wr[k];
        xr[c] = acc;
    }
}

// backward_indexes = inverse permutation (argsort of a permutation)
__global__ void bwd_kernel(const int* __restrict__ fwd, float* __restrict__ out)
{
    int e = blockIdx.x * 256 + threadIdx.x;
    if (e >= Tfull * Bc) return;
    int i = e / Bc, b = e % Bc;
    int t = fwd[e];
    out[(size_t)t * Bc + b] = (float)i;              // int values, stored in f32 d_out tail
}

// ---------------------------------------------------------------- launch
extern "C" void kernel_launch(void* const* d_in, const int* in_sizes, int n_in,
                              void* d_out, int out_size, void* d_ws, size_t ws_size,
                              hipStream_t stream)
{
    (void)in_sizes; (void)n_in; (void)out_size; (void)ws_size;
    const float* img    = (const float*)d_in[0];
    const int*   fwd    = (const int*)  d_in[1];
    const float* conv_w = (const float*)d_in[2];
    const float* conv_b = (const float*)d_in[3];
    const float* pos    = (const float*)d_in[4];
    const float* cls    = (const float*)d_in[5];
    const float* ln1g   = (const float*)d_in[6];
    const float* ln1b   = (const float*)d_in[7];
    const float* qkvw   = (const float*)d_in[8];
    const float* projw  = (const float*)d_in[9];
    const float* projb  = (const float*)d_in[10];
    const float* ln2g   = (const float*)d_in[11];
    const float* ln2b   = (const float*)d_in[12];
    const float* gatew  = (const float*)d_in[13];
    const float* gateb  = (const float*)d_in[14];
    const float* w1     = (const float*)d_in[15];
    const float* b1     = (const float*)d_in[16];
    const float* w2     = (const float*)d_in[17];
    const float* b2     = (const float*)d_in[18];
    const float* lnfg   = (const float*)d_in[19];
    const float* lnfb   = (const float*)d_in[20];

    char* ws = (char*)d_ws;
    float*          x      = (float*)ws;          ws += (size_t)NTOK * Cc * 4;
    unsigned short* hbf    = (unsigned short*)ws; ws += (size_t)NTOK * Cc * 2;
    float*          qkvbuf = (float*)ws;          ws += (size_t)NTOK * 3 * Cc * 4;
    unsigned short* obf    = (unsigned short*)ws; ws += (size_t)NTOK * Cc * 2;
    unsigned short* Hbf    = (unsigned short*)ws; ws += (size_t)NTOK * HIDc * 2;
    float*          webuf  = (float*)ws;          ws += (size_t)NTOK * Ee * 4;

    float* feat = (float*)d_out;                   // [50,128,384] then [196,128]

    patchify_kernel<<<Bc * Tt, 256, 0, stream>>>(img, fwd, conv_w, conv_b, pos, cls, x);
    bwd_kernel<<<(Tfull * Bc + 255) / 256, 256, 0, stream>>>(fwd, feat + (size_t)Tt * Bc * Cc);

    for (int l = 0; l < 12; ++l) {
        ln_kernel<<<NTOK / 8, 256, 0, stream>>>(x, ln1g + l * Cc, ln1b + l * Cc, hbf);
        gemm_wmma<M_STORE><<<dim3(3 * Cc / 64, NTOK / 128), 256, 0, stream>>>(
            hbf, qkvw + (size_t)l * Cc * 3 * Cc, nullptr, qkvbuf, nullptr, nullptr, Cc, 3 * Cc);
        attn_kernel<<<Bc * NHh, 256, 0, stream>>>(qkvbuf, obf);
        gemm_wmma<M_ACC_BIAS><<<dim3(Cc / 64, NTOK / 128), 256, 0, stream>>>(
            obf, projw + (size_t)l * Cc * Cc, projb + (size_t)l * Cc, x, nullptr, nullptr, Cc, Cc);
        ln_kernel<<<NTOK / 8, 256, 0, stream>>>(x, ln2g + l * Cc, ln2b + l * Cc, hbf);
        gate_kernel<<<NTOK / 256, 256, 0, stream>>>(
            hbf, gatew + (size_t)l * Cc * Ee, gateb + (size_t)l * Ee, webuf);
        for (int e = 0; e < Ee; ++e) {
            gemm_wmma<M_GELU_BF16><<<dim3(HIDc / 64, NTOK / 128), 256, 0, stream>>>(
                hbf, w1 + ((size_t)l * Ee + e) * Cc * HIDc,
                b1 + ((size_t)l * Ee + e) * HIDc, nullptr, Hbf, nullptr, Cc, HIDc);
            gemm_wmma<M_ACC_BIAS_SCALE><<<dim3(Cc / 64, NTOK / 128), 256, 0, stream>>>(
                Hbf, w2 + ((size_t)l * Ee + e) * HIDc * Cc,
                b2 + ((size_t)l * Ee + e) * Cc, x, nullptr, webuf + e, HIDc, Cc);
        }
    }
    lnf_kernel<<<NTOK / 8, 256, 0, stream>>>(x, lnfg, lnfb, feat);
}
